// Attention_16200616640614
// MI455X (gfx1250) — matrix operand order repaired and checked by reading, not compiled, not run on hardware
//
#include <hip/hip_runtime.h>

// ---------------------------------------------------------------------------
// Fused causal MHA forward for MI455X (gfx1250, wave32, WMMA bf16 16x16x32)
// B=2, S=2048, D=1024, H=16, hd=64
// Round 2: + Tensor Data Mover (tensor_load_to_lds) for Q/K tile staging in
// the attention kernel (s_wait_tensorcnt sync), + global prefetch in GEMMs.
// ---------------------------------------------------------------------------

typedef __attribute__((ext_vector_type(16))) __bf16 v16bf;
typedef __attribute__((ext_vector_type(8)))  float  v8f;
typedef __attribute__((ext_vector_type(4)))  unsigned int v4u;
typedef __attribute__((ext_vector_type(8)))  int    v8i;
typedef __attribute__((ext_vector_type(4)))  int    v4i;

#define BB 2
#define SSQ 2048
#define DDIM 1024
#define HH 16
#define HD 64
#define A_ELEMS  (BB * SSQ * DDIM)       // 4194304 (a output)
#define KV_ELEMS (BB * HH * SSQ * HD)    // 4194304 (one of k/v in present)

#if __has_builtin(__builtin_amdgcn_tensor_load_to_lds) && \
    __has_builtin(__builtin_amdgcn_s_wait_tensorcnt)
#define HAVE_TDM 1
#if __has_include(<hip/amd_detail/amd_gfx1250_TDM.h>)
// therock-10.0 headers -> 6-arg builtin (extra int32x8 group before cpol)
#define TDM_LOAD(g0, g1, g2, g3) \
    __builtin_amdgcn_tensor_load_to_lds((g0), (g1), (g2), (g3), \
                                        (v8i){0,0,0,0,0,0,0,0}, 0)
#else
// ROCm 7.2 (clang-22) -> 5-arg builtin
#define TDM_LOAD(g0, g1, g2, g3) \
    __builtin_amdgcn_tensor_load_to_lds((g0), (g1), (g2), (g3), 0)
#endif
#endif

__device__ __forceinline__ unsigned short f2bf(float x) {
    unsigned u = __builtin_bit_cast(unsigned, x);
    u += 0x7FFFu + ((u >> 16) & 1u);     // round-to-nearest-even
    return (unsigned short)(u >> 16);
}

union FragU { unsigned u[8]; v16bf v; };

// A-matrix 16x32 bf16 fragment from row-major LDS [row][k] (ISA 7.12.2):
// lanes 0-15: m=lane, V0..3 -> K 0..7, V4..7 -> K 16..23
// lanes 16-31: m=lane-16, V0..3 -> K 8..15, V4..7 -> K 24..31
__device__ __forceinline__ v16bf frag_a16(const unsigned short* lds, int row0,
                                          int stride, int koff) {
    const int lane = threadIdx.x & 31;
    const int g = lane >> 4, m = lane & 15;
    const unsigned short* base = lds + (row0 + m) * stride + koff + (g ? 8 : 0);
    FragU r;
#pragma unroll
    for (int j = 0; j < 8; ++j) {
        int k = ((j >= 4) ? 16 : 0) + 2 * (j & 3);
        r.u[j] = *(const unsigned*)(base + k);
    }
    return r.v;
}

// B-matrix 32x16 bf16 fragment. LDS holds B transposed: [n][k] row-major,
// so the (K=2j, 2j+1) pair for a fixed column n is one aligned dword.
// lanes 0-15: n=lane, V0..3 -> K 0..7, V4..7 -> K 16..23 ; lanes 16-31: n=lane-16, K += 8
__device__ __forceinline__ v16bf frag_b16(const unsigned short* lds, int row0,
                                          int stride, int koff) {
    const int lane = threadIdx.x & 31;
    const int g = lane >> 4, n = lane & 15;
    const unsigned short* base = lds + (row0 + n) * stride + koff + (g ? 8 : 0);
    FragU r;
#pragma unroll
    for (int j = 0; j < 8; ++j)
        r.u[j] = *(const unsigned*)(base + ((j >= 4) ? 16 : 0) + 2 * (j & 3));
    return r.v;
}

__device__ __forceinline__ v8f wmma_bf16(v16bf a, v16bf b, v8f c) {
    return __builtin_amdgcn_wmma_f32_16x16x32_bf16(false, a, false, b,
                                                   (short)0, c, false, false);
}

__device__ __forceinline__ v8f v8zero() {
    v8f z;
#pragma unroll
    for (int j = 0; j < 8; ++j) z[j] = 0.0f;
    return z;
}

#ifdef HAVE_TDM
// TDM: 2D bf16 tile load global->LDS. Tile = tile_rows x 64 bf16, global row
// stride 64 elements (contiguous), LDS row stride 72 bf16 (pad 4 dwords after
// every 32 dwords). One issue per wave; EXEC ignored by TDM.
__device__ __forceinline__ void tdm_load_rows64_bf16(unsigned lds_off,
                                                     const unsigned short* gptr,
                                                     int tile_rows) {
    unsigned long long ga = (unsigned long long)(size_t)gptr;
    v4u g0;
    g0[0] = 1u;                                        // count=1, user mode
    g0[1] = lds_off;                                   // LDS byte address
    g0[2] = (unsigned)(ga & 0xFFFFFFFFull);            // global addr [31:0]
    g0[3] = (unsigned)((ga >> 32) & 0x01FFFFFFull)     // global addr [56:32]
          | (2u << 30);                                // type=2 ("image")
    v8i g1;
    // data_size=1(2B) | pad_enable | pad_interval=4(32dw) | pad_amount=3(4dw)
    g1[0] = (int)(0x00010000u | 0x00100000u | (4u << 22) | (3u << 25));
    g1[1] = (int)(64u << 16);                          // tensor_dim0=64 (lo16)
    g1[2] = (int)((unsigned)(tile_rows & 0xFFFF) << 16); // dim0 hi=0 | dim1 lo
    g1[3] = (int)(64u << 16);                          // dim1 hi=0 | tile_dim0=64
    g1[4] = (int)(tile_rows & 0xFFFF);                 // tile_dim1 | tile_dim2=0
    g1[5] = 64;                                        // dim0_stride lo32
    g1[6] = 0;                                         // stride hi | dim1_stride lo
    g1[7] = 0;
    v4i z4 = {0, 0, 0, 0};
    TDM_LOAD(g0, g1, z4, z4);
}
#endif

// ---------------------------------------------------------------------------
// Kernel 1: qkv = x @ W_attn + b_attn  (4096x1024 @ 1024x3072)
// Writes Q/K/V bf16 [B,H,S,hd] to ws; K/V f32 to the `present` part of d_out.
// Block tile 128x128, 8 waves (4x2) of 32x64, K-step 32.
// ---------------------------------------------------------------------------
__global__ __launch_bounds__(256)
void qkv_kernel(const float* __restrict__ X, const float* __restrict__ W,
                const float* __restrict__ bias,
                unsigned short* __restrict__ Qbf, unsigned short* __restrict__ Kbf,
                unsigned short* __restrict__ Vbf, float* __restrict__ out) {
    __shared__ unsigned short As[128 * 40];   // [row][k], stride 40
    __shared__ unsigned short Bs[128 * 40];   // [n][k] (B transposed)
    const int tid  = threadIdx.x;
    const int wave = tid >> 5;
    const int m0 = blockIdx.y * 128;
    const int n0 = blockIdx.x * 128;
    const int wm = (wave & 3) * 32;
    const int wn = (wave >> 2) * 64;

    v8f acc[2][4];
#pragma unroll
    for (int i = 0; i < 2; ++i)
#pragma unroll
        for (int j = 0; j < 4; ++j) acc[i][j] = v8zero();

    for (int k0 = 0; k0 < 1024; k0 += 32) {
        __syncthreads();
#pragma unroll
        for (int i = 0; i < 16; ++i) {
            int e = tid + i * 256;            // 4096 elements each
            int r = e >> 5, c = e & 31;
            As[r * 40 + c] = f2bf(X[(m0 + r) * 1024 + k0 + c]);
            int kr = e >> 7, nc = e & 127;
            Bs[nc * 40 + kr] = f2bf(W[(k0 + kr) * 3072 + n0 + nc]);
        }
        if (k0 + 32 < 1024) {                 // prefetch next K-step tiles
            __builtin_prefetch(&X[(m0 + (tid >> 1)) * 1024 + k0 + 32 +
                                  ((tid & 1) << 4)], 0, 1);
            if (tid < 128)
                __builtin_prefetch(&W[(k0 + 32 + (tid >> 2)) * 3072 + n0 +
                                      ((tid & 3) << 5)], 0, 1);
        }
        __syncthreads();
        v16bf a0 = frag_a16(As, wm,      40, 0);
        v16bf a1 = frag_a16(As, wm + 16, 40, 0);
#pragma unroll
        for (int nt = 0; nt < 4; ++nt) {
            v16bf b = frag_b16(Bs, wn + nt * 16, 40, 0);
            acc[0][nt] = wmma_bf16(a0, b, acc[0][nt]);
            acc[1][nt] = wmma_bf16(a1, b, acc[1][nt]);
        }
    }

    const int lane = tid & 31;
    const int g = lane >> 4, nl = lane & 15;
#pragma unroll
    for (int mi = 0; mi < 2; ++mi)
#pragma unroll
        for (int nt = 0; nt < 4; ++nt)
#pragma unroll
            for (int j = 0; j < 8; ++j) {
                int row = m0 + wm + mi * 16 + j + 8 * g;
                int col = n0 + wn + nt * 16 + nl;
                float v = acc[mi][nt][j] + bias[col];
                int which = col >> 10;            // 0=q 1=k 2=v
                int d  = col & 1023;
                int h  = d >> 6, dd = d & 63;
                int b  = row >> 11, s = row & 2047;
                int idx = ((b * HH + h) * SSQ + s) * HD + dd;
                if (which == 0) {
                    Qbf[idx] = f2bf(v);
                } else if (which == 1) {
                    Kbf[idx] = f2bf(v);
                    out[A_ELEMS + idx] = v;                 // present[0] = k
                } else {
                    Vbf[idx] = f2bf(v);
                    out[A_ELEMS + KV_ELEMS + idx] = v;      // present[1] = v
                }
            }
}

// ---------------------------------------------------------------------------
// Kernel 2: flash attention. Block = 128 q rows (8 waves x 16 rows), 64-key
// tiles, online softmax, P staged through per-wave LDS to feed P@V WMMAs.
// Q and K tiles staged via the Tensor Data Mover when available.
// ---------------------------------------------------------------------------
__global__ __launch_bounds__(256)
void attn_kernel(const unsigned short* __restrict__ Qbf,
                 const unsigned short* __restrict__ Kbf,
                 const unsigned short* __restrict__ Vbf,
                 unsigned short* __restrict__ attn) {
    __shared__ unsigned short Qs[128 * 72];      // [qrow][dd]
    __shared__ unsigned short Ks[64 * 72];       // [key][dd]
    __shared__ unsigned short Vs[64 * 72];       // [dd][key]  (transposed)
    __shared__ unsigned short Ps[8 * 16 * 72];   // per-wave P [qrow][key]

    const int tid  = threadIdx.x;
    const int wave = tid >> 5;
    const int lane = tid & 31;
    const int g = lane >> 4, nl = lane & 15;
    const int q0 = blockIdx.x * 128;
    const int bh = blockIdx.y;

    const unsigned* Qg = (const unsigned*)Qbf;   // dword views (2 bf16/dword)
    const unsigned* Kg = (const unsigned*)Kbf;
    const unsigned* Vg = (const unsigned*)Vbf;
    (void)Qg; (void)Kg;

#ifdef HAVE_TDM
    if (wave == 0) {
        tdm_load_rows64_bf16((unsigned)(size_t)&Qs[0],
                             Qbf + (bh * SSQ + q0) * HD, 128);
        __builtin_amdgcn_s_wait_tensorcnt(0);
    }
#else
#pragma unroll
    for (int i = 0; i < 16; ++i) {               // 4096 dwords of Q tile
        int e = tid + i * 256;
        int r = e >> 5, cd = e & 31;
        *(unsigned*)&Qs[r * 72 + 2 * cd] = Qg[(bh * SSQ + q0 + r) * 32 + cd];
    }
#endif
    __syncthreads();
    v16bf qf0 = frag_a16(Qs, wave * 16, 72, 0);
    v16bf qf1 = frag_a16(Qs, wave * 16, 72, 32);

    v8f o[4];
    float mrow[8], lrow[8];
#pragma unroll
    for (int t = 0; t < 4; ++t) o[t] = v8zero();
#pragma unroll
    for (int j = 0; j < 8; ++j) { mrow[j] = -3.0e38f; lrow[j] = 0.0f; }

    const int nkt = q0 / 64 + 2;                 // causal upper bound in tiles
    for (int kt = 0; kt < nkt; ++kt) {
        const int k0 = kt * 64;
        __syncthreads();
#ifdef HAVE_TDM
        if (wave == 0)                            // async K DMA overlaps V copy
            tdm_load_rows64_bf16((unsigned)(size_t)&Ks[0],
                                 Kbf + (bh * SSQ + k0) * HD, 64);
#endif
#pragma unroll
        for (int i = 0; i < 8; ++i) {            // V transpose copy (+K fallback)
            int e = tid + i * 256;
            int r = e >> 5, cd = e & 31;
#ifndef HAVE_TDM
            *(unsigned*)&Ks[r * 72 + 2 * cd] = Kg[(bh * SSQ + k0 + r) * 32 + cd];
#endif
            unsigned vv = Vg[(bh * SSQ + k0 + r) * 32 + cd];
            Vs[(2 * cd)     * 72 + r] = (unsigned short)(vv & 0xFFFFu);
            Vs[(2 * cd + 1) * 72 + r] = (unsigned short)(vv >> 16);
        }
#ifdef HAVE_TDM
        if (wave == 0) __builtin_amdgcn_s_wait_tensorcnt(0);
#endif
        __syncthreads();

        v8f s[4];
#pragma unroll
        for (int nt = 0; nt < 4; ++nt) {
            v8f z = v8zero();
            z = wmma_bf16(qf0, frag_b16(Ks, nt * 16, 72, 0),  z);
            z = wmma_bf16(qf1, frag_b16(Ks, nt * 16, 72, 32), z);
            s[nt] = z;
        }

        // scale + causal mask + per-row max of this tile
        float mt[8];
#pragma unroll
        for (int j = 0; j < 8; ++j) mt[j] = -3.0e38f;
#pragma unroll
        for (int nt = 0; nt < 4; ++nt)
#pragma unroll
            for (int j = 0; j < 8; ++j) {
                int rowg = q0 + wave * 16 + j + 8 * g;
                int colg = k0 + nt * 16 + nl;
                float val = s[nt][j] * 0.125f;   // 1/sqrt(64)
                if (colg > rowg) val = -1.0e10f; // matches reference -NEG
                s[nt][j] = val;
                mt[j] = fmaxf(mt[j], val);
            }
#pragma unroll
        for (int j = 0; j < 8; ++j)
#pragma unroll
            for (int off = 1; off < 16; off <<= 1)
                mt[j] = fmaxf(mt[j], __shfl_xor(mt[j], off, 32));

        float fj[8];
#pragma unroll
        for (int j = 0; j < 8; ++j) {
            float mnew = fmaxf(mrow[j], mt[j]);
            fj[j] = __expf(mrow[j] - mnew);
            mrow[j] = mnew;
        }

        float rs[8];
#pragma unroll
        for (int j = 0; j < 8; ++j) rs[j] = 0.0f;
#pragma unroll
        for (int nt = 0; nt < 4; ++nt)
#pragma unroll
            for (int j = 0; j < 8; ++j) {
                float p = __expf(s[nt][j] - mrow[j]);
                s[nt][j] = p;
                rs[j] += p;
            }
#pragma unroll
        for (int j = 0; j < 8; ++j) {
#pragma unroll
            for (int off = 1; off < 16; off <<= 1)
                rs[j] += __shfl_xor(rs[j], off, 32);
            lrow[j] = lrow[j] * fj[j] + rs[j];
        }
#pragma unroll
        for (int dt = 0; dt < 4; ++dt)
#pragma unroll
            for (int j = 0; j < 8; ++j) o[dt][j] *= fj[j];

        // C-layout -> A-fragment reshape of P via per-wave LDS buffer
        unsigned short* Pw = Ps + wave * 16 * 72;
#pragma unroll
        for (int nt = 0; nt < 4; ++nt)
#pragma unroll
            for (int j = 0; j < 8; ++j)
                Pw[(j + 8 * g) * 72 + nt * 16 + nl] = f2bf(s[nt][j]);
        // in-wave LDS store->load turnaround: drain DS counter explicitly
        asm volatile("s_wait_dscnt 0" ::: "memory");
        v16bf pa0 = frag_a16(Pw, 0, 72, 0);
        v16bf pa1 = frag_a16(Pw, 0, 72, 32);
#pragma unroll
        for (int dt = 0; dt < 4; ++dt) {
            o[dt] = wmma_bf16(pa0, frag_b16(Vs, dt * 16, 72, 0),  o[dt]);
            o[dt] = wmma_bf16(pa1, frag_b16(Vs, dt * 16, 72, 32), o[dt]);
        }
    }

    const int b = bh / HH, h = bh % HH;
#pragma unroll
    for (int dt = 0; dt < 4; ++dt)
#pragma unroll
        for (int j = 0; j < 8; ++j) {
            int rowg = b * SSQ + q0 + wave * 16 + j + 8 * g;
            int col  = h * HD + dt * 16 + nl;
            attn[rowg * DDIM + col] = f2bf(o[dt][j] / lrow[j]);
        }
}

// ---------------------------------------------------------------------------
// Kernel 3: a = attn @ W_proj + b_proj  (4096x1024 @ 1024x1024, f32 out)
// ---------------------------------------------------------------------------
__global__ __launch_bounds__(256)
void proj_kernel(const unsigned short* __restrict__ Abf,
                 const float* __restrict__ W, const float* __restrict__ bias,
                 float* __restrict__ out) {
    __shared__ unsigned short As[128 * 40];
    __shared__ unsigned short Bs[128 * 40];
    const int tid  = threadIdx.x;
    const int wave = tid >> 5;
    const int m0 = blockIdx.y * 128;
    const int n0 = blockIdx.x * 128;
    const int wm = (wave & 3) * 32;
    const int wn = (wave >> 2) * 64;
    const unsigned* Ag = (const unsigned*)Abf;

    v8f acc[2][4];
#pragma unroll
    for (int i = 0; i < 2; ++i)
#pragma unroll
        for (int j = 0; j < 4; ++j) acc[i][j] = v8zero();

    for (int k0 = 0; k0 < 1024; k0 += 32) {
        __syncthreads();
#pragma unroll
        for (int i = 0; i < 8; ++i) {            // A tile: 2048 dwords (bf16)
            int e = tid + i * 256;
            int r = e >> 4, cd = e & 15;
            *(unsigned*)&As[r * 40 + 2 * cd] = Ag[(m0 + r) * 512 + (k0 >> 1) + cd];
        }
#pragma unroll
        for (int i = 0; i < 16; ++i) {           // W tile: 4096 f32 -> bf16
            int e = tid + i * 256;
            int kr = e >> 7, nc = e & 127;
            Bs[nc * 40 + kr] = f2bf(W[(k0 + kr) * 1024 + n0 + nc]);
        }
        if (k0 + 32 < 1024 && tid < 128)         // prefetch next W tile
            __builtin_prefetch(&W[(k0 + 32 + (tid >> 2)) * 1024 + n0 +
                                  ((tid & 3) << 5)], 0, 1);
        __syncthreads();
        v16bf a0 = frag_a16(As, wm,      40, 0);
        v16bf a1 = frag_a16(As, wm + 16, 40, 0);
#pragma unroll
        for (int nt = 0; nt < 4; ++nt) {
            v16bf b = frag_b16(Bs, wn + nt * 16, 40, 0);
            acc[0][nt] = wmma_bf16(a0, b, acc[0][nt]);
            acc[1][nt] = wmma_bf16(a1, b, acc[1][nt]);
        }
    }

    const int lane = tid & 31;
    const int g = lane >> 4, nl = lane & 15;
#pragma unroll
    for (int mi = 0; mi < 2; ++mi)
#pragma unroll
        for (int nt = 0; nt < 4; ++nt)
#pragma unroll
            for (int j = 0; j < 8; ++j) {
                int row = m0 + wm + mi * 16 + j + 8 * g;
                int col = n0 + wn + nt * 16 + nl;
                out[row * 1024 + col] = acc[mi][nt][j] + bias[col];
            }
}

// ---------------------------------------------------------------------------
extern "C" void kernel_launch(void* const* d_in, const int* in_sizes, int n_in,
                              void* d_out, int out_size, void* d_ws, size_t ws_size,
                              hipStream_t stream) {
    const float* x      = (const float*)d_in[0];
    const float* W_attn = (const float*)d_in[1];
    const float* b_attn = (const float*)d_in[2];
    const float* W_proj = (const float*)d_in[3];
    const float* b_proj = (const float*)d_in[4];
    float* out = (float*)d_out;

    // workspace: Qbf | Kbf | Vbf | attn-out, all bf16 (8 MB each, 32 MB total)
    unsigned short* Qbf  = (unsigned short*)d_ws;
    unsigned short* Kbf  = Qbf + KV_ELEMS;
    unsigned short* Vbf  = Kbf + KV_ELEMS;
    unsigned short* attn = Vbf + KV_ELEMS;

    qkv_kernel <<<dim3(24, 32), 256, 0, stream>>>(x, W_attn, b_attn,
                                                  Qbf, Kbf, Vbf, out);
    attn_kernel<<<dim3(16, 32), 256, 0, stream>>>(Qbf, Kbf, Vbf, attn);
    proj_kernel<<<dim3(8, 32), 256, 0, stream>>>(attn, W_proj, b_proj, out);
}
